// RNNGenerator_33157147525733
// MI455X (gfx1250) — matrix operand order repaired and checked by reading, not compiled, not run on hardware
//
#include <hip/hip_runtime.h>
#include <hip/hip_bf16.h>

// ---------------------------------------------------------------------------
// Bidirectional projected LSTM for MI455X (gfx1250, wave32, WMMA).
//
// Strategy (latency-bound sequential scan):
//  1) pack_kernel:     permute/transpose/pad weights to bf16 operand layouts.
//  2) input_gemm:      Gpre[dir][t] = x_t @ Wx + b  for all t in parallel
//                      (bf16 WMMA, f32 accum, stored in WMMA D-fragment layout).
//  3) recurrent:       2 persistent workgroups (one per direction), 32 waves,
//                      full 320KB LDS residency for Wm^T / P^T / m / H;
//                      c lives in registers; per step: gate WMMA (+Gpre as C),
//                      in-wave ds_swizzle quad exchange for i/j/f/o, then
//                      projection WMMA; 2 barriers per step.
// Precision: bf16 matrix operands, f32 accumulation (chip-rewarded path).
// ---------------------------------------------------------------------------

typedef __attribute__((ext_vector_type(16))) __bf16 v16bf;
typedef __attribute__((ext_vector_type(8)))  __bf16 v8bf;
typedef __attribute__((ext_vector_type(8)))  float  v8f;
typedef __attribute__((ext_vector_type(4)))  float  v4f;
typedef __attribute__((ext_vector_type(4)))  unsigned int v4u;

#define B_   128
#define T_   400
#define I_   100
#define S_   100
#define F_   200
#define NG   400          // 4*S gate columns (permuted n = 4*s + g)
#define KM   224          // recurrent K (200 -> pad 224, 7 chunks of 32)
#define KMC  7
#define NP   208          // projection N (200 -> pad 208, 13 tiles)
#define KP   128          // projection K (100 -> pad 128, 4 chunks)
#define KPC  4
#define KX   128          // input K (100 -> pad 128, 4 chunks)
#define KXC  4
#define GT_TILES 200      // 8 M-tiles x 25 N-tiles
#define PJ_TILES 104      // 8 M-tiles x 13 N-tiles

#if __has_builtin(__builtin_amdgcn_exp2f)
#define EXP2F __builtin_amdgcn_exp2f
#else
#define EXP2F exp2f
#endif
#if __has_builtin(__builtin_amdgcn_rcpf)
#define RCPF __builtin_amdgcn_rcpf
#else
#define RCPF(x) (1.0f / (x))
#endif

__device__ __forceinline__ float fsig(float x) {
  return RCPF(1.0f + EXP2F(-1.4426950408889634f * x));
}
__device__ __forceinline__ float ftanh(float x) {
  float e = EXP2F(2.8853900817779268f * x);
  return 1.0f - 2.0f * RCPF(e + 1.0f);
}
__device__ __forceinline__ __bf16 f2bf(float f) {
  unsigned u = __builtin_bit_cast(unsigned, f);
  unsigned r = (u + 0x7FFFu + ((u >> 16) & 1u)) >> 16;
  return __builtin_bit_cast(__bf16, (unsigned short)r);
}
__device__ __forceinline__ float i2f(int x) { return __builtin_bit_cast(float, x); }

// permuted gate column n = 4*s + g  ->  original W column g*100 + s
__device__ __forceinline__ int colmap(int n) { return (n & 3) * 100 + (n >> 2); }

// A-fragment (16x32 bf16, M x K) from row-major LDS buffer [rows][stride].
// lane L: M = L&15 ; group g = L>>4 ; VGPR0-3: K = k0+8g..+7 ; VGPR4-7: K = k0+16+8g..+7
__device__ __forceinline__ v16bf lds_load_a16(const __bf16* base, int stride,
                                              int row0, int k0, int lane) {
  const __bf16* p = base + (size_t)(row0 + (lane & 15)) * stride + k0 + ((lane >> 4) << 3);
  v8bf r0 = *(const v8bf*)p;
  v8bf r1 = *(const v8bf*)(p + 16);
  return __builtin_shufflevector(r0, r1, 0,1,2,3,4,5,6,7,8,9,10,11,12,13,14,15);
}

// B-fragment (32x16 bf16, K x N) from N-major (transposed) LDS buffer [N][stride=Kpad].
// lane L: N = L&15 ; g = L>>4 ; VGPR0-3: K = k0+8g..+7 ; VGPR4-7: K = k0+16+8g..+7
__device__ __forceinline__ v16bf lds_load_b16(const __bf16* baseT, int stride,
                                              int col0, int k0, int lane) {
  const __bf16* p = baseT + (size_t)(col0 + (lane & 15)) * stride + k0 + ((lane >> 4) << 3);
  v8bf r0 = *(const v8bf*)p;
  v8bf r1 = *(const v8bf*)(p + 16);
  return __builtin_shufflevector(r0, r1, 0,1,2,3,4,5,6,7,8,9,10,11,12,13,14,15);
}

// ---------------------------------------------------------------------------
// Kernel 1: pack weights into bf16 padded/permuted operand layouts in ws.
//   WmtG[dir][n in 0..399][k in 0..223]  = W[100+k][colmap(n)]   (0 pad k>=200)
//   PtG [dir][n in 0..207][k in 0..127]  = P[k][n]               (0 pad)
//   WxtG[dir][n in 0..399][k in 0..127]  = W[k][colmap(n)]       (0 pad k>=100)
// ---------------------------------------------------------------------------
#define WMT_ELEMS (2 * NG * KM)   // 179200
#define PT_ELEMS  (2 * NP * KP)   //  53248
#define WXT_ELEMS (2 * NG * KX)   // 102400

__global__ void lstm_pack_kernel(const float* __restrict__ Wf, const float* __restrict__ Wb,
                                 const float* __restrict__ Pf, const float* __restrict__ Pb,
                                 __bf16* __restrict__ WmtG, __bf16* __restrict__ PtG,
                                 __bf16* __restrict__ WxtG) {
  int idx = blockIdx.x * blockDim.x + threadIdx.x;
  if (idx < WMT_ELEMS) {
    int dir = idx / (NG * KM), r = idx % (NG * KM);
    int n = r / KM, k = r % KM;
    const float* W = dir ? Wb : Wf;
    float v = (k < 200) ? W[(size_t)(100 + k) * NG + colmap(n)] : 0.0f;
    WmtG[idx] = f2bf(v);
  } else if (idx < WMT_ELEMS + PT_ELEMS) {
    int j = idx - WMT_ELEMS;
    int dir = j / (NP * KP), r = j % (NP * KP);
    int n = r / KP, k = r % KP;
    const float* P = dir ? Pb : Pf;
    float v = (n < 200 && k < 100) ? P[(size_t)k * F_ + n] : 0.0f;
    PtG[j] = f2bf(v);
  } else if (idx < WMT_ELEMS + PT_ELEMS + WXT_ELEMS) {
    int j = idx - WMT_ELEMS - PT_ELEMS;
    int dir = j / (NG * KX), r = j % (NG * KX);
    int n = r / KX, k = r % KX;
    const float* W = dir ? Wb : Wf;
    float v = (k < 100) ? W[(size_t)k * NG + colmap(n)] : 0.0f;
    WxtG[j] = f2bf(v);
  }
}

// ---------------------------------------------------------------------------
// Kernel 2: Gpre[dir][t][tile][lane][0..7] = (x_t @ Wx)+bias  (f32 D-fragment
// layout so the scan can load C with two b128 loads). Grid (T, 2), 1024 thr.
// ---------------------------------------------------------------------------
__global__ void lstm_input_gemm(const float* __restrict__ z, const __bf16* __restrict__ WxtG,
                                const float* __restrict__ bf_, const float* __restrict__ bb_,
                                float* __restrict__ Gpre) {
  extern __shared__ unsigned char smem[];
  __bf16* Wxt = (__bf16*)smem;                       // [NG][KX]   102400 B
  __bf16* Xt  = (__bf16*)(smem + NG * KX * 2);       // [128][KX]   32768 B
  const int t = blockIdx.x, dir = blockIdx.y;
  const int tid = threadIdx.x;

  { // stage Wx^T (coalesced 16B chunks)
    const v4u* src = (const v4u*)(WxtG + (size_t)dir * NG * KX);
    v4u* dst = (v4u*)Wxt;
    for (int i = tid; i < NG * KX * 2 / 16; i += 1024) dst[i] = src[i];
  }
  for (int i = tid; i < B_ * KX; i += 1024) { // stage x_t (f32 -> bf16, zero pad)
    int b = i >> 7, k = i & (KX - 1);
    float v = (k < I_) ? z[(size_t)b * (T_ * I_) + (size_t)t * I_ + k] : 0.0f;
    Xt[i] = f2bf(v);
  }
  __syncthreads();

  const int lane = tid & 31, wave = tid >> 5;
  const float* bias = dir ? bb_ : bf_;
  for (int tile = wave; tile < GT_TILES; tile += 32) {
    int Mt = tile / 25, Nt = tile % 25;
    v8f acc = {};
#pragma unroll
    for (int kc = 0; kc < KXC; ++kc) {
      v16bf a = lds_load_a16(Xt, KX, Mt * 16, kc * 32, lane);
      v16bf b = lds_load_b16(Wxt, KX, Nt * 16, kc * 32, lane);
      acc = __builtin_amdgcn_wmma_f32_16x16x32_bf16(false, a, false, b, (short)0, acc,
                                                    false, false);
    }
    int n = Nt * 16 + (lane & 15);
    float bv = bias[colmap(n)];
#pragma unroll
    for (int v = 0; v < 8; ++v) acc[v] += bv;
    float* gp = Gpre + ((((size_t)dir * T_ + t) * GT_TILES + tile) * 32 + lane) * 8;
    *(v4f*)gp       = v4f{acc[0], acc[1], acc[2], acc[3]};
    *(v4f*)(gp + 4) = v4f{acc[4], acc[5], acc[6], acc[7]};
  }
}

// ---------------------------------------------------------------------------
// Kernel 3: persistent recurrent scan. Grid = 2 (dir), 1024 threads = 32 waves.
// LDS: Wm^T 179200 + P^T 53248 + Mbuf 57344 + Hbuf 32768 = 322560 B (<=320KB).
// ---------------------------------------------------------------------------
#define LDS_WMT 0
#define LDS_PT  (NG * KM * 2)                 // 179200
#define LDS_MB  (LDS_PT + NP * KP * 2)        // 232448
#define LDS_HB  (LDS_MB + B_ * KM * 2)        // 289792
#define LDS_TOT (LDS_HB + B_ * KP * 2)        // 322560

__global__ void lstm_recurrent(const float* __restrict__ cf0, const float* __restrict__ mf0,
                               const float* __restrict__ cb0, const float* __restrict__ mb0,
                               const __bf16* __restrict__ WmtG, const __bf16* __restrict__ PtG,
                               const float* __restrict__ Gpre, float* __restrict__ out) {
  extern __shared__ unsigned char smem[];
  __bf16* Wmt = (__bf16*)(smem + LDS_WMT);   // [NG][KM]
  __bf16* Pt  = (__bf16*)(smem + LDS_PT);    // [NP][KP]
  __bf16* Mb  = (__bf16*)(smem + LDS_MB);    // [128][KM]  recurrent state m (bf16)
  __bf16* Hb  = (__bf16*)(smem + LDS_HB);    // [128][KP]  hidden h (bf16)

  const int dir = blockIdx.x;
  const int tid = threadIdx.x, lane = tid & 31, wave = tid >> 5;

  { // stage Wm^T and P^T once
    const v4u* src = (const v4u*)(WmtG + (size_t)dir * NG * KM);
    v4u* dst = (v4u*)Wmt;
    for (int i = tid; i < NG * KM * 2 / 16; i += 1024) dst[i] = src[i];
    const v4u* src2 = (const v4u*)(PtG + (size_t)dir * NP * KP);
    v4u* dst2 = (v4u*)Pt;
    for (int i = tid; i < NP * KP * 2 / 16; i += 1024) dst2[i] = src2[i];
  }
  const float* m0 = dir ? mb0 : mf0;
  for (int i = tid; i < B_ * KM; i += 1024) { // m_0 -> bf16, zero pad (NaN-safe)
    int b = i / KM, k = i % KM;
    Mb[i] = f2bf((k < F_) ? m0[(size_t)b * F_ + k] : 0.0f);
  }
  { // zero Hbuf (pad columns must be finite-zero forever)
    v4u zero = {};
    v4u* h = (v4u*)Hb;
    for (int i = tid; i < B_ * KP * 2 / 16; i += 1024) h[i] = zero;
  }

  // cell state c in registers of owning lanes (col n=4s+g, owner computes redundantly)
  const float* c0 = dir ? cb0 : cf0;
  float creg[7][8];
#pragma unroll
  for (int it = 0; it < 7; ++it) {
    int tile = wave + it * 32;
    if (tile < GT_TILES) {
      int Mt = tile / 25, Nt = tile % 25;
      int s = Nt * 4 + ((lane & 15) >> 2);
      int brow = Mt * 16 + 8 * (lane >> 4);
#pragma unroll
      for (int v = 0; v < 8; ++v) creg[it][v] = c0[(size_t)(brow + v) * S_ + s];
    }
  }
  __syncthreads();

  const float* gbase = Gpre + (size_t)dir * T_ * GT_TILES * 256;

  for (int tau = 0; tau < T_; ++tau) {
    const int t = dir ? (T_ - 1 - tau) : tau;
    // ---------------- gate GEMM + LSTM cell update ----------------
#pragma unroll
    for (int it = 0; it < 7; ++it) {
      int tile = wave + it * 32;
      if (tile < GT_TILES) {
        int Mt = tile / 25, Nt = tile % 25;
        const float* gp = gbase + ((size_t)t * GT_TILES + tile) * 256 + lane * 8;
        v4f g0 = *(const v4f*)gp;
        v4f g1 = *(const v4f*)(gp + 4);
        v8f acc;
        acc[0] = g0[0]; acc[1] = g0[1]; acc[2] = g0[2]; acc[3] = g0[3];
        acc[4] = g1[0]; acc[5] = g1[1]; acc[6] = g1[2]; acc[7] = g1[3];
        if (tau + 1 < T_) { // prefetch next timestep's C-fragment into caches
          int tn = dir ? (t - 1) : (t + 1);
          __builtin_prefetch(gbase + ((size_t)tn * GT_TILES + tile) * 256 + lane * 8, 0, 1);
        }
#pragma unroll
        for (int kc = 0; kc < KMC; ++kc) {
          v16bf a = lds_load_a16(Mb, KM, Mt * 16, kc * 32, lane);
          v16bf b = lds_load_b16(Wmt, KM, Nt * 16, kc * 32, lane);
          acc = __builtin_amdgcn_wmma_f32_16x16x32_bf16(false, a, false, b, (short)0, acc,
                                                        false, false);
        }
        // lanes 4q..4q+3 hold (i,j,f,o) of s = Nt*4+q; quad-broadcast via ds_swizzle
        int brow = Mt * 16 + 8 * (lane >> 4);
        int s = Nt * 4 + ((lane & 15) >> 2);
        bool writer = ((lane & 3) == 0);
#pragma unroll
        for (int v = 0; v < 8; ++v) {
          int bits = __builtin_bit_cast(int, acc[v]);
          float gi = i2f(__builtin_amdgcn_ds_swizzle(bits, 0x8000)); // quad lane 0: i
          float gj = i2f(__builtin_amdgcn_ds_swizzle(bits, 0x8055)); // quad lane 1: j
          float gf = i2f(__builtin_amdgcn_ds_swizzle(bits, 0x80AA)); // quad lane 2: f
          float go = i2f(__builtin_amdgcn_ds_swizzle(bits, 0x80FF)); // quad lane 3: o
          float cn = fsig(gf + 1.0f) * creg[it][v] + fsig(gi) * ftanh(gj);
          creg[it][v] = cn;
          float h = fsig(go) * ftanh(cn);
          if (writer) Hb[(size_t)(brow + v) * KP + s] = f2bf(h);
        }
      }
    }
    __syncthreads();
    // ---------------- projection GEMM: m = H @ P ----------------
#pragma unroll
    for (int it = 0; it < 4; ++it) {
      int tile = wave + it * 32;
      if (tile < PJ_TILES) {
        int Mt = tile / 13, Nt = tile % 13;
        v8f acc = {};
#pragma unroll
        for (int kc = 0; kc < KPC; ++kc) {
          v16bf a = lds_load_a16(Hb, KP, Mt * 16, kc * 32, lane);
          v16bf b = lds_load_b16(Pt, KP, Nt * 16, kc * 32, lane);
          acc = __builtin_amdgcn_wmma_f32_16x16x32_bf16(false, a, false, b, (short)0, acc,
                                                        false, false);
        }
        int n = Nt * 16 + (lane & 15);
        int brow = Mt * 16 + 8 * (lane >> 4);
#pragma unroll
        for (int v = 0; v < 8; ++v) {
          Mb[(size_t)(brow + v) * KM + n] = f2bf(acc[v]); // pad cols write 0 (P pad = 0)
          if (n < F_) {
            // out[b, t*2F + dir*F + n]
            out[(size_t)(brow + v) * (T_ * 2 * F_) + (size_t)t * (2 * F_) + dir * F_ + n] =
                acc[v];
          }
        }
      }
    }
    __syncthreads();
  }
}

// ---------------------------------------------------------------------------
// Host launcher. Inputs (setup_inputs order):
//  0 z, 1 c_f0, 2 m_f0, 3 c_b0, 4 m_b0, 5 W_f, 6 b_f, 7 P_f, 8 W_b, 9 b_b, 10 P_b
// ws layout: WmtG | PtG | WxtG | Gpre  (~164.5 MB)
// ---------------------------------------------------------------------------
extern "C" void kernel_launch(void* const* d_in, const int* in_sizes, int n_in,
                              void* d_out, int out_size, void* d_ws, size_t ws_size,
                              hipStream_t stream) {
  (void)in_sizes; (void)n_in; (void)out_size; (void)ws_size;
  const float* z   = (const float*)d_in[0];
  const float* cf0 = (const float*)d_in[1];
  const float* mf0 = (const float*)d_in[2];
  const float* cb0 = (const float*)d_in[3];
  const float* mb0 = (const float*)d_in[4];
  const float* Wf  = (const float*)d_in[5];
  const float* bf_ = (const float*)d_in[6];
  const float* Pf  = (const float*)d_in[7];
  const float* Wb  = (const float*)d_in[8];
  const float* bb_ = (const float*)d_in[9];
  const float* Pb  = (const float*)d_in[10];
  float* out = (float*)d_out;

  unsigned char* ws = (unsigned char*)d_ws;
  const size_t off_wmt  = 0;
  const size_t off_pt   = off_wmt + (size_t)WMT_ELEMS * 2;   // 358400
  const size_t off_wxt  = off_pt  + (size_t)PT_ELEMS  * 2;   // 464896
  const size_t off_gpre = off_wxt + (size_t)WXT_ELEMS * 2;   // 669696
  __bf16* WmtG = (__bf16*)(ws + off_wmt);
  __bf16* PtG  = (__bf16*)(ws + off_pt);
  __bf16* WxtG = (__bf16*)(ws + off_wxt);
  float*  Gpre = (float*)(ws + off_gpre);                    // 163,840,000 B

  // allow >64KB dynamic LDS (gfx1250 WGP has 320KB)
  (void)hipFuncSetAttribute((const void*)lstm_input_gemm,
                            hipFuncAttributeMaxDynamicSharedMemorySize,
                            NG * KX * 2 + B_ * KX * 2);
  (void)hipFuncSetAttribute((const void*)lstm_recurrent,
                            hipFuncAttributeMaxDynamicSharedMemorySize, LDS_TOT);

  const int pack_total = WMT_ELEMS + PT_ELEMS + WXT_ELEMS;
  lstm_pack_kernel<<<(pack_total + 255) / 256, 256, 0, stream>>>(Wf, Wb, Pf, Pb,
                                                                 WmtG, PtG, WxtG);
  lstm_input_gemm<<<dim3(T_, 2), 1024, NG * KX * 2 + B_ * KX * 2, stream>>>(
      z, WxtG, bf_, bb_, Gpre);
  lstm_recurrent<<<2, 1024, LDS_TOT, stream>>>(cf0, mf0, cb0, mb0, WmtG, PtG, Gpre, out);
}
